// WrapMemRNN_84189948936536
// MI455X (gfx1250) — matrix operand order repaired and checked by reading, not compiled, not run on hardware
//
#include <hip/hip_runtime.h>
#include <hip/hip_bf16.h>

// ---------------------------------------------------------------- constants
#define HSZ   512
#define H3    1536
#define VSZ   32000
#define BSZ   32
#define SSZ   2048
#define QSZ   64
#define NHOPS 3
#define LDSR  544   // LDS row stride in bf16 elements (512 + 32 pad)

typedef __bf16 bf16_t;
typedef __attribute__((ext_vector_type(16))) __bf16 v16bf;
typedef __attribute__((ext_vector_type(8)))  __bf16 v8bf;
typedef __attribute__((ext_vector_type(8)))  float  v8f;

// ---------------------------------------------------------------- helpers
__device__ __forceinline__ bf16_t f32_to_bf16(float f) {
  unsigned u = __builtin_bit_cast(unsigned, f);
  u += 0x7FFFu + ((u >> 16) & 1u);           // round-to-nearest-even
  return __builtin_bit_cast(bf16_t, (unsigned short)(u >> 16));
}
__device__ __forceinline__ float sigmoidf_(float x) { return 1.0f / (1.0f + expf(-x)); }

__device__ __forceinline__ v8f wmma_bf16(v16bf a, v16bf b, v8f c) {
  return __builtin_amdgcn_wmma_f32_16x16x32_bf16(false, a, false, b, (short)0, c,
                                                 false, false);
}

// A fragment (16x32 bf16, M x K). Per ISA: lanes 0-15 hold K={0..7,16..23},
// lanes 16-31 hold K={8..15,24..31}, 2 values per VGPR. `row` points at A[m][0].
__device__ __forceinline__ v16bf load_a_frag(const bf16_t* row, int kt, int lane) {
  int base = kt * 32 + ((lane >> 4) ? 8 : 0);
  v8bf lo = *(const v8bf*)(row + base);
  v8bf hi = *(const v8bf*)(row + base + 16);
  return __builtin_shufflevector(lo, hi, 0,1,2,3,4,5,6,7,8,9,10,11,12,13,14,15);
}

// B fragment (32x16 bf16, K x N) from B^T stored [N, ldb]. lane%16 = N,
// lanes 0-15 hold K={0..7,16..23}, lanes 16-31 hold K={8..15,24..31}.
__device__ __forceinline__ v16bf load_b_frag(const bf16_t* bt, int ldb, int n,
                                             int kt, int lane) {
  const bf16_t* p = bt + (size_t)n * ldb + kt * 32 + ((lane >> 4) ? 8 : 0);
  v8bf lo = *(const v8bf*)p;
  v8bf hi = *(const v8bf*)(p + 16);
  return __builtin_shufflevector(lo, hi, 0,1,2,3,4,5,6,7,8,9,10,11,12,13,14,15);
}

// ---------------------------------------------------------------- prep kernels
__global__ void k_transpose_cast(const float* __restrict__ in, bf16_t* __restrict__ out,
                                 int K, int N) {  // in [K,N] f32 -> out [N,K] bf16
  int i = blockIdx.x * blockDim.x + threadIdx.x;
  if (i >= N * K) return;
  int n = i / K, k = i % K;
  out[i] = f32_to_bf16(in[(size_t)k * N + n]);
}

__global__ void k_gather_embed(const int* __restrict__ idx, const float* __restrict__ embed,
                               bf16_t* __restrict__ x, int rows) {
  int i = blockIdx.x * blockDim.x + threadIdx.x;
  if (i >= rows * HSZ) return;
  int r = i >> 9, h = i & (HSZ - 1);
  x[i] = f32_to_bf16(embed[(size_t)idx[r] * HSZ + h]);
}

// ---------------------------------------------------------------- generic GEMM
// C[M,N] f32 = A[M,K] bf16 @ B^T[N,K] bf16 (+ bias[N]).  grid = (M/16, N/64), 32 thr.
__global__ void k_gemm_bf16(const bf16_t* __restrict__ A, const bf16_t* __restrict__ BT,
                            const float* __restrict__ bias, float* __restrict__ C,
                            int N, int K) {
  const int lane = threadIdx.x & 31;
  const int mt = blockIdx.x;
  const int n0 = blockIdx.y * 64;
  v8f acc[4] = {};
  const bf16_t* arow = A + (size_t)(mt * 16 + (lane & 15)) * K;
  const int kts = K >> 5;
  for (int kt = 0; kt < kts; ++kt) {
    v16bf a = load_a_frag(arow, kt, lane);
#pragma unroll
    for (int t = 0; t < 4; ++t) {
      v16bf b = load_b_frag(BT, K, n0 + t * 16 + (lane & 15), kt, lane);
      acc[t] = wmma_bf16(a, b, acc[t]);
    }
  }
  const int mrow = mt * 16 + 8 * (lane >> 4);
  const int ncol = lane & 15;
#pragma unroll
  for (int t = 0; t < 4; ++t) {
    int n = n0 + t * 16 + ncol;
    float bb = bias ? bias[n] : 0.0f;
#pragma unroll
    for (int v = 0; v < 8; ++v)
      C[(size_t)(mrow + v) * N + n] = acc[t][v] + bb;
  }
}

// ---------------------------------------------------------------- GRU scan
// Persistent 1-WG kernel, 32 waves. Wave w owns hidden columns [16w,16w+16).
// xW: [B*T, 3H] f32 (bi already folded), Wh^T: [3H, H] bf16, out: [B*T, H] f32.
__global__ __launch_bounds__(1024) void k_gru_scan(const float* __restrict__ xW,
                                                   const bf16_t* __restrict__ WhT,
                                                   const float* __restrict__ bh,
                                                   float* __restrict__ out,
                                                   int T, int rev) {
  __shared__ __align__(32) bf16_t hbuf[32 * LDSR];
  const int tid = threadIdx.x, lane = tid & 31, wv = tid >> 5;
  for (int i = tid; i < 32 * LDSR; i += 1024)
    hbuf[i] = __builtin_bit_cast(bf16_t, (unsigned short)0);
  float hreg[2][8];
#pragma unroll
  for (int mt = 0; mt < 2; ++mt)
#pragma unroll
    for (int v = 0; v < 8; ++v) hreg[mt][v] = 0.0f;
  __syncthreads();

  const int jcol = wv * 16 + (lane & 15);
  const float bhr = bh[jcol], bhz = bh[HSZ + jcol], bhn = bh[2 * HSZ + jcol];

  for (int step = 0; step < T; ++step) {
    const int t = rev ? (T - 1 - step) : step;
    v8f acc[2][3] = {};
    const bf16_t* arow0 = &hbuf[(size_t)(lane & 15) * LDSR];
    const bf16_t* arow1 = &hbuf[(size_t)(16 + (lane & 15)) * LDSR];
#pragma unroll 4
    for (int kt = 0; kt < 16; ++kt) {
      v16bf a0 = load_a_frag(arow0, kt, lane);
      v16bf a1 = load_a_frag(arow1, kt, lane);
#pragma unroll
      for (int g = 0; g < 3; ++g) {
        v16bf b = load_b_frag(WhT, HSZ, g * HSZ + wv * 16 + (lane & 15), kt, lane);
        acc[0][g] = wmma_bf16(a0, b, acc[0][g]);
        acc[1][g] = wmma_bf16(a1, b, acc[1][g]);
      }
    }
    __syncthreads();  // all reads of hbuf done
#pragma unroll
    for (int mt = 0; mt < 2; ++mt) {
#pragma unroll
      for (int v = 0; v < 8; ++v) {
        int b = mt * 16 + 8 * (lane >> 4) + v;
        const float* xwr = xW + ((size_t)b * T + t) * H3;
        float r = sigmoidf_(xwr[jcol] + acc[mt][0][v] + bhr);
        float z = sigmoidf_(xwr[HSZ + jcol] + acc[mt][1][v] + bhz);
        float n = tanhf(xwr[2 * HSZ + jcol] + acc[mt][2][v] + bhn);
        float h = (1.0f - z) * n + z * hreg[mt][v];
        hreg[mt][v] = h;
        hbuf[(size_t)b * LDSR + jcol] = f32_to_bf16(h);
        out[((size_t)b * T + t) * HSZ + jcol] = h;
      }
    }
    __syncthreads();  // new state visible
  }
}

// ---------------------------------------------------------------- facts / q
__global__ void k_facts_combine(float* __restrict__ of, const float* __restrict__ ob,
                                bf16_t* __restrict__ fbf, int n) {
  int i = blockIdx.x * blockDim.x + threadIdx.x;
  if (i >= n) return;
  float f = of[i] + ob[i];
  of[i] = f;
  fbf[i] = f32_to_bf16(f);
}

__global__ void k_extract_q(const float* __restrict__ oq, float* __restrict__ q,
                            float* __restrict__ mem0) {
  int i = blockIdx.x * blockDim.x + threadIdx.x;
  if (i >= BSZ * HSZ) return;
  int b = i >> 9, h = i & (HSZ - 1);
  float v = oq[((size_t)b * QSZ + (QSZ - 1)) * HSZ + h];
  q[i] = v;
  mem0[i] = v;
}

// ---------------------------------------------------------------- attention
// Fused: z-features built on the fly; epilogue does tanh(c+ba)*Wb row-reduction
// into g_acc[B*S] with atomics. grid = (B*S/16, H/64), 32 thr.
__global__ void k_attn(const float* __restrict__ facts, const float* __restrict__ q,
                       const float* __restrict__ mem, const bf16_t* __restrict__ WaT,
                       const float* __restrict__ ba, const float* __restrict__ Wb,
                       float* __restrict__ g_acc) {
  const int lane = threadIdx.x & 31;
  const int mt = blockIdx.x;
  const int n0 = blockIdx.y * 64;
  const int marow = mt * 16 + (lane & 15);
  const int b = marow >> 11;  // S = 2048, 16-row tiles never straddle batches
  const float* frow = facts + (size_t)marow * HSZ;
  const float* qrow = q + (size_t)b * HSZ;
  const float* mrow = mem + (size_t)b * HSZ;

  v8f acc[4] = {};
  for (int kt = 0; kt < 64; ++kt) {
    v16bf a;
    const int base = kt * 32 + ((lane >> 4) ? 8 : 0);
#pragma unroll
    for (int j = 0; j < 16; ++j) {
      int k = base + (j < 8 ? j : j + 8);
      int sec = k >> 9, kk = k & (HSZ - 1);
      float f = frow[kk];
      float val;
      if (sec == 0)      val = f * qrow[kk];
      else if (sec == 1) val = f * mrow[kk];
      else if (sec == 2) val = fabsf(f - qrow[kk]);
      else               val = fabsf(f - mrow[kk]);
      a[j] = f32_to_bf16(val);
    }
#pragma unroll
    for (int t = 0; t < 4; ++t) {
      v16bf bf = load_b_frag(WaT, 4 * HSZ, n0 + t * 16 + (lane & 15), kt, lane);
      acc[t] = wmma_bf16(a, bf, acc[t]);
    }
  }
  // epilogue: partial_m = sum_n tanh(c + ba[n]) * Wb[n]
  float part[8];
#pragma unroll
  for (int v = 0; v < 8; ++v) part[v] = 0.0f;
  const int ncol = lane & 15;
#pragma unroll
  for (int t = 0; t < 4; ++t) {
    int n = n0 + t * 16 + ncol;
    float ban = ba[n], wbn = Wb[n];
#pragma unroll
    for (int v = 0; v < 8; ++v) part[v] += tanhf(acc[t][v] + ban) * wbn;
  }
#pragma unroll
  for (int off = 1; off < 16; off <<= 1)
#pragma unroll
    for (int v = 0; v < 8; ++v) part[v] += __shfl_xor(part[v], off, 32);
  if ((lane & 15) == 0) {
    int mbase = mt * 16 + 8 * (lane >> 4);
#pragma unroll
    for (int v = 0; v < 8; ++v) atomicAdd(&g_acc[mbase + v], part[v]);
  }
}

__global__ void k_gfin(float* __restrict__ g, const float* __restrict__ gacc,
                       const float* __restrict__ bb, int n) {
  int i = blockIdx.x * blockDim.x + threadIdx.x;
  if (i >= n) return;
  g[i] = sigmoidf_(gacc[i] + bb[0]);
}

// ---------------------------------------------------------------- episode scan
__global__ __launch_bounds__(1024) void k_episode(const float* __restrict__ fr,
                                                  const float* __restrict__ fh,
                                                  const float* __restrict__ g,
                                                  const bf16_t* __restrict__ UrT,
                                                  const bf16_t* __restrict__ UT,
                                                  const float* __restrict__ bur,
                                                  const float* __restrict__ bu,
                                                  float* __restrict__ e) {
  __shared__ __align__(32) bf16_t cbuf[32 * LDSR];
  const int tid = threadIdx.x, lane = tid & 31, wv = tid >> 5;
  for (int i = tid; i < 32 * LDSR; i += 1024)
    cbuf[i] = __builtin_bit_cast(bf16_t, (unsigned short)0);
  float creg[2][8];
#pragma unroll
  for (int mt = 0; mt < 2; ++mt)
#pragma unroll
    for (int v = 0; v < 8; ++v) creg[mt][v] = 0.0f;
  __syncthreads();

  const int jcol = wv * 16 + (lane & 15);
  const float burj = bur[jcol], buj = bu[jcol];

  for (int t = 0; t < SSZ; ++t) {
    v8f aur[2] = {}, au[2] = {};
    const bf16_t* arow0 = &cbuf[(size_t)(lane & 15) * LDSR];
    const bf16_t* arow1 = &cbuf[(size_t)(16 + (lane & 15)) * LDSR];
#pragma unroll 4
    for (int kt = 0; kt < 16; ++kt) {
      v16bf a0 = load_a_frag(arow0, kt, lane);
      v16bf a1 = load_a_frag(arow1, kt, lane);
      v16bf br = load_b_frag(UrT, HSZ, wv * 16 + (lane & 15), kt, lane);
      v16bf bw = load_b_frag(UT,  HSZ, wv * 16 + (lane & 15), kt, lane);
      aur[0] = wmma_bf16(a0, br, aur[0]);
      aur[1] = wmma_bf16(a1, br, aur[1]);
      au[0]  = wmma_bf16(a0, bw, au[0]);
      au[1]  = wmma_bf16(a1, bw, au[1]);
    }
    __syncthreads();
#pragma unroll
    for (int mt = 0; mt < 2; ++mt) {
#pragma unroll
      for (int v = 0; v < 8; ++v) {
        int b = mt * 16 + 8 * (lane >> 4) + v;
        size_t row = (size_t)b * SSZ + t;
        float r  = sigmoidf_(fr[row * HSZ + jcol] + aur[mt][v] + burj);
        float ht = tanhf(fh[row * HSZ + jcol] + r * (au[mt][v] + buj));
        float gt = g[row];
        float c  = gt * ht + (1.0f - gt) * creg[mt][v];
        creg[mt][v] = c;
        cbuf[(size_t)b * LDSR + jcol] = f32_to_bf16(c);
      }
    }
    __syncthreads();
  }
#pragma unroll
  for (int mt = 0; mt < 2; ++mt)
#pragma unroll
    for (int v = 0; v < 8; ++v) {
      int b = mt * 16 + 8 * (lane >> 4) + v;
      e[(size_t)b * HSZ + jcol] = creg[mt][v];
    }
}

// ---------------------------------------------------------------- small GEMMs
__global__ void k_mem_update(const float* __restrict__ mem_in, const float* __restrict__ e,
                             const float* __restrict__ q, const float* __restrict__ Wm,
                             const float* __restrict__ bm, float* __restrict__ mem_out) {
  int i = blockIdx.x * blockDim.x + threadIdx.x;
  if (i >= BSZ * HSZ) return;
  int b = i >> 9, n = i & (HSZ - 1);
  const float* mi = mem_in + (size_t)b * HSZ;
  const float* ei = e + (size_t)b * HSZ;
  const float* qi = q + (size_t)b * HSZ;
  float acc = bm[n];
  for (int k = 0; k < HSZ; ++k) acc += mi[k] * Wm[(size_t)k * HSZ + n];
  for (int k = 0; k < HSZ; ++k) acc += ei[k] * Wm[(size_t)(k + HSZ) * HSZ + n];
  for (int k = 0; k < HSZ; ++k) acc += qi[k] * Wm[(size_t)(k + 2 * HSZ) * HSZ + n];
  mem_out[i] = acc > 0.0f ? acc : 0.0f;
}

__global__ void k_logits(const float* __restrict__ mem, const float* __restrict__ q,
                         const float* __restrict__ Wo, const float* __restrict__ bo,
                         float* __restrict__ out) {
  int i = blockIdx.x * blockDim.x + threadIdx.x;
  if (i >= BSZ * VSZ) return;
  int b = i / VSZ, v = i % VSZ;
  const float* mi = mem + (size_t)b * HSZ;
  const float* qi = q + (size_t)b * HSZ;
  float acc = bo[v];
  for (int k = 0; k < HSZ; ++k) acc += mi[k] * Wo[(size_t)k * VSZ + v];
  for (int k = 0; k < HSZ; ++k) acc += qi[k] * Wo[(size_t)(k + HSZ) * VSZ + v];
  out[i] = acc;
}

// ---------------------------------------------------------------- host side
extern "C" void kernel_launch(void* const* d_in, const int* in_sizes, int n_in,
                              void* d_out, int out_size, void* d_ws, size_t ws_size,
                              hipStream_t stream) {
  const int*   src      = (const int*)d_in[0];
  const int*   question = (const int*)d_in[1];
  const float* embed    = (const float*)d_in[2];
  const float* Wi_f = (const float*)d_in[3],  *Wh_f = (const float*)d_in[4];
  const float* bi_f = (const float*)d_in[5],  *bh_f = (const float*)d_in[6];
  const float* Wi_b = (const float*)d_in[7],  *Wh_b = (const float*)d_in[8];
  const float* bi_b = (const float*)d_in[9],  *bh_b = (const float*)d_in[10];
  const float* Wi_q = (const float*)d_in[11], *Wh_q = (const float*)d_in[12];
  const float* bi_q = (const float*)d_in[13], *bh_q = (const float*)d_in[14];
  const float* Wa = (const float*)d_in[15], *ba = (const float*)d_in[16];
  const float* Wb = (const float*)d_in[17], *bb = (const float*)d_in[18];
  const float* Wr = (const float*)d_in[19], *br = (const float*)d_in[20];
  const float* Ur = (const float*)d_in[21], *bur = (const float*)d_in[22];
  const float* Wg = (const float*)d_in[23], *bg = (const float*)d_in[24];
  const float* U  = (const float*)d_in[25], *bu = (const float*)d_in[26];
  const float* Wm = (const float*)d_in[27], *bm = (const float*)d_in[28];
  const float* Wo = (const float*)d_in[29], *bo = (const float*)d_in[30];

  // workspace carve-out (256 B aligned)
  char* base = (char*)d_ws;
  size_t off = 0;
  auto alloc = [&](size_t bytes) -> void* {
    off = (off + 255) & ~(size_t)255;
    void* p = base + off;
    off += bytes;
    return p;
  };
  const size_t MBS = (size_t)BSZ * SSZ;          // 65536 rows
  bf16_t* x_bf   = (bf16_t*)alloc(MBS * HSZ * 2);
  bf16_t* xq_bf  = (bf16_t*)alloc((size_t)BSZ * QSZ * HSZ * 2);
  float*  xW     = (float*) alloc(MBS * H3 * 4);          // shared by f / b passes
  float*  xWq    = (float*) alloc((size_t)BSZ * QSZ * H3 * 4);
  float*  of     = (float*) alloc(MBS * HSZ * 4);         // later holds facts f32
  float*  ob     = (float*) alloc(MBS * HSZ * 4);
  float*  oq     = (float*) alloc((size_t)BSZ * QSZ * HSZ * 4);
  bf16_t* fct_bf = (bf16_t*)alloc(MBS * HSZ * 2);
  float*  fr     = (float*) alloc(MBS * HSZ * 4);
  float*  fh     = (float*) alloc(MBS * HSZ * 4);
  float*  g_acc  = (float*) alloc(MBS * 4);
  float*  gbuf   = (float*) alloc(MBS * 4);
  float*  qbuf   = (float*) alloc((size_t)BSZ * HSZ * 4);
  float*  mem_a  = (float*) alloc((size_t)BSZ * HSZ * 4);
  float*  mem_b  = (float*) alloc((size_t)BSZ * HSZ * 4);
  float*  ebuf   = (float*) alloc((size_t)BSZ * HSZ * 4);
  bf16_t* WiTf = (bf16_t*)alloc((size_t)H3 * HSZ * 2);
  bf16_t* WiTb = (bf16_t*)alloc((size_t)H3 * HSZ * 2);
  bf16_t* WiTq = (bf16_t*)alloc((size_t)H3 * HSZ * 2);
  bf16_t* WhTf = (bf16_t*)alloc((size_t)H3 * HSZ * 2);
  bf16_t* WhTb = (bf16_t*)alloc((size_t)H3 * HSZ * 2);
  bf16_t* WhTq = (bf16_t*)alloc((size_t)H3 * HSZ * 2);
  bf16_t* WaT  = (bf16_t*)alloc((size_t)4 * HSZ * HSZ * 2);
  bf16_t* WrT  = (bf16_t*)alloc((size_t)HSZ * HSZ * 2);
  bf16_t* WgT  = (bf16_t*)alloc((size_t)HSZ * HSZ * 2);
  bf16_t* UrT  = (bf16_t*)alloc((size_t)HSZ * HSZ * 2);
  bf16_t* UT   = (bf16_t*)alloc((size_t)HSZ * HSZ * 2);
  (void)ws_size; (void)in_sizes; (void)n_in; (void)out_size;

  // --- weight prep: transpose + cast to bf16 ([K,N] f32 -> [N,K] bf16)
  auto tc = [&](const float* in, bf16_t* out, int K, int N) {
    int tot = K * N;
    k_transpose_cast<<<(tot + 255) / 256, 256, 0, stream>>>(in, out, K, N);
  };
  tc(Wi_f, WiTf, HSZ, H3);  tc(Wi_b, WiTb, HSZ, H3);  tc(Wi_q, WiTq, HSZ, H3);
  tc(Wh_f, WhTf, HSZ, H3);  tc(Wh_b, WhTb, HSZ, H3);  tc(Wh_q, WhTq, HSZ, H3);
  tc(Wa, WaT, 4 * HSZ, HSZ);
  tc(Wr, WrT, HSZ, HSZ);  tc(Wg, WgT, HSZ, HSZ);
  tc(Ur, UrT, HSZ, HSZ);  tc(U,  UT,  HSZ, HSZ);

  // --- embedding gather (bf16)
  k_gather_embed<<<(int)((MBS * HSZ + 255) / 256), 256, 0, stream>>>(src, embed, x_bf, (int)MBS);
  k_gather_embed<<<(BSZ * QSZ * HSZ + 255) / 256, 256, 0, stream>>>(question, embed, xq_bf,
                                                                    BSZ * QSZ);

  // --- forward GRU
  k_gemm_bf16<<<dim3(MBS / 16, H3 / 64), 32, 0, stream>>>(x_bf, WiTf, bi_f, xW, H3, HSZ);
  k_gru_scan<<<1, 1024, 0, stream>>>(xW, WhTf, bh_f, of, SSZ, 0);
  // --- backward GRU (reuses xW buffer)
  k_gemm_bf16<<<dim3(MBS / 16, H3 / 64), 32, 0, stream>>>(x_bf, WiTb, bi_b, xW, H3, HSZ);
  k_gru_scan<<<1, 1024, 0, stream>>>(xW, WhTb, bh_b, ob, SSZ, 1);
  // --- question GRU
  k_gemm_bf16<<<dim3((BSZ * QSZ) / 16, H3 / 64), 32, 0, stream>>>(xq_bf, WiTq, bi_q, xWq,
                                                                  H3, HSZ);
  k_gru_scan<<<1, 1024, 0, stream>>>(xWq, WhTq, bh_q, oq, QSZ, 0);

  // --- facts = of + ob (f32 in `of`, bf16 copy), fact projections
  k_facts_combine<<<(int)((MBS * HSZ + 255) / 256), 256, 0, stream>>>(of, ob, fct_bf,
                                                                      (int)(MBS * HSZ));
  k_gemm_bf16<<<dim3(MBS / 16, HSZ / 64), 32, 0, stream>>>(fct_bf, WrT, br, fr, HSZ, HSZ);
  k_gemm_bf16<<<dim3(MBS / 16, HSZ / 64), 32, 0, stream>>>(fct_bf, WgT, bg, fh, HSZ, HSZ);
  k_extract_q<<<(BSZ * HSZ + 255) / 256, 256, 0, stream>>>(oq, qbuf, mem_a);

  // --- episodic memory hops
  float* mem_cur = mem_a;
  float* mem_nxt = mem_b;
  for (int hop = 0; hop < NHOPS; ++hop) {
    hipMemsetAsync(g_acc, 0, MBS * 4, stream);
    k_attn<<<dim3(MBS / 16, HSZ / 64), 32, 0, stream>>>(of, qbuf, mem_cur, WaT, ba, Wb,
                                                        g_acc);
    k_gfin<<<(int)((MBS + 255) / 256), 256, 0, stream>>>(gbuf, g_acc, bb, (int)MBS);
    k_episode<<<1, 1024, 0, stream>>>(fr, fh, gbuf, UrT, UT, bur, bu, ebuf);
    k_mem_update<<<(BSZ * HSZ + 255) / 256, 256, 0, stream>>>(mem_cur, ebuf, qbuf, Wm, bm,
                                                              mem_nxt);
    float* t = mem_cur; mem_cur = mem_nxt; mem_nxt = t;
  }

  // --- logits
  k_logits<<<(BSZ * VSZ + 255) / 256, 256, 0, stream>>>(mem_cur, qbuf, Wo, bo,
                                                        (float*)d_out);
}
